// SelfAttention_17179869189
// MI455X (gfx1250) — matrix operand order repaired and checked by reading, not compiled, not run on hardware
//
#include <hip/hip_runtime.h>

#define B_ 8
#define C_ 256
#define N_ 2048

typedef __attribute__((ext_vector_type(16))) _Float16 v16h;
typedef __attribute__((ext_vector_type(8)))  _Float16 v8h;
typedef __attribute__((ext_vector_type(8)))  float    v8f;

union AFrag { v16h v; v8h h[2]; };

__device__ __forceinline__ v8f vzero8() {
  v8f z;
  #pragma unroll
  for (int j = 0; j < 8; ++j) z[j] = 0.0f;
  return z;
}

// ---------------------------------------------------------------------------
// Kernel 0a: convert weights to f16; fold temperature 1/sqrt(256)=1/16 into Wq
// ---------------------------------------------------------------------------
__global__ void convert_w_kernel(const float* __restrict__ Wq,
                                 const float* __restrict__ Wk,
                                 const float* __restrict__ Wv,
                                 _Float16* __restrict__ Wqh,
                                 _Float16* __restrict__ Wkh,
                                 _Float16* __restrict__ Wvh) {
  int i = blockIdx.x * blockDim.x + threadIdx.x;
  if (i < C_ * C_) {
    Wqh[i] = (_Float16)(Wq[i] * 0.0625f);
    Wkh[i] = (_Float16)Wk[i];
    Wvh[i] = (_Float16)Wv[i];
  }
}

// ---------------------------------------------------------------------------
// Kernel 0b: x [B,C,N] f32 -> xT [B,N,C] f16 (LDS-tiled transpose)
// ---------------------------------------------------------------------------
__global__ void transpose_x_kernel(const float* __restrict__ x,
                                   _Float16* __restrict__ xT) {
  __shared__ float tile[32][33];
  int b  = blockIdx.z;
  int n0 = blockIdx.x * 32;
  int c0 = blockIdx.y * 32;
  int tx = threadIdx.x, ty = threadIdx.y;   // block (32, 8)
  const float* xp = x + ((size_t)b * C_ + c0) * N_ + n0;
  #pragma unroll
  for (int i = 0; i < 32; i += 8)
    tile[ty + i][tx] = xp[(size_t)(ty + i) * N_ + tx];
  __syncthreads();
  _Float16* op = xT + ((size_t)b * N_ + n0) * C_ + c0;
  #pragma unroll
  for (int i = 0; i < 32; i += 8)
    op[(size_t)(ty + i) * C_ + tx] = (_Float16)tile[tx][ty + i];
}

// ---------------------------------------------------------------------------
// Kernel 1: Q/K projection. One wave owns a 16-row n-tile of one output.
// Out[n, co] = sum_ci xT[n, ci] * W[co, ci]   (A = xT tile, B = W^T tile)
// ---------------------------------------------------------------------------
__global__ __launch_bounds__(256) void proj_qk_kernel(
    const _Float16* __restrict__ xT,
    const _Float16* __restrict__ Wqh,
    const _Float16* __restrict__ Wkh,
    _Float16* __restrict__ Qt,
    _Float16* __restrict__ Kt) {
  int lane = threadIdx.x & 31;
  int wv   = threadIdx.x >> 5;
  int l16  = lane & 15;
  int h    = lane >> 4;

  int w   = blockIdx.x * 8 + wv;     // 0..2047
  int mat = w >> 10;                 // 0 = Q, 1 = K
  int r   = w & 1023;
  int b   = r >> 7;
  int nt  = r & 127;

  const _Float16* W   = mat ? Wkh : Wqh;
  _Float16*       Out = mat ? Kt  : Qt;

  v8f acc[16];
  #pragma unroll
  for (int t = 0; t < 16; ++t) acc[t] = vzero8();

  // A-frag lane row: m = l16; K runs [c0+h*8 .. +7] and [c0+16+h*8 .. +7]
  const _Float16* xrow = xT + ((size_t)b * N_ + nt * 16 + l16) * C_;

  #pragma unroll
  for (int kc = 0; kc < 8; ++kc) {
    int c0 = kc * 32;
    AFrag a;
    a.h[0] = *(const v8h*)(xrow + c0 + h * 8);
    a.h[1] = *(const v8h*)(xrow + c0 + 16 + h * 8);
    #pragma unroll
    for (int ct = 0; ct < 16; ++ct) {
      // B-frag lane col: co = ct*16 + l16; K runs [c0+h*8 .. +7] and [c0+16+h*8 .. +7]
      const _Float16* wrow = W + (size_t)(ct * 16 + l16) * C_ + c0 + h * 8;
      AFrag bf;
      bf.h[0] = *(const v8h*)(wrow);
      bf.h[1] = *(const v8h*)(wrow + 16);
      acc[ct] = __builtin_amdgcn_wmma_f32_16x16x32_f16(
          false, a.v, false, bf.v, (short)0, acc[ct], false, false);
    }
  }

  _Float16* orow0 = Out + ((size_t)b * N_ + nt * 16) * C_;
  #pragma unroll
  for (int ct = 0; ct < 16; ++ct)
    #pragma unroll
    for (int v = 0; v < 8; ++v)
      orow0[(size_t)(h * 8 + v) * C_ + ct * 16 + l16] = (_Float16)acc[ct][v];
}

// ---------------------------------------------------------------------------
// Kernel 2: V projection with swapped roles so V lands in [C, N] layout.
// Vc[co, n] = sum_ci W[co, ci] * xT[n, ci]   (A = W tile, B = xT^T tile)
// ---------------------------------------------------------------------------
__global__ __launch_bounds__(256) void proj_v_kernel(
    const _Float16* __restrict__ xT,
    const _Float16* __restrict__ Wvh,
    _Float16* __restrict__ Vc) {
  int lane = threadIdx.x & 31;
  int wv   = threadIdx.x >> 5;
  int l16  = lane & 15;
  int h    = lane >> 4;

  int w  = blockIdx.x * 8 + wv;   // 0..1023
  int b  = w >> 7;
  int nt = w & 127;

  v8f acc[16];
  #pragma unroll
  for (int t = 0; t < 16; ++t) acc[t] = vzero8();

  // B-frag lane col: n = nt*16 + l16; K runs [c0+h*8 .. +7] and [c0+16+h*8 .. +7] along xT row
  const _Float16* xrow = xT + ((size_t)b * N_ + nt * 16 + l16) * C_;

  #pragma unroll
  for (int kc = 0; kc < 8; ++kc) {
    int c0 = kc * 32;
    AFrag bf;
    bf.h[0] = *(const v8h*)(xrow + c0 + h * 8);
    bf.h[1] = *(const v8h*)(xrow + c0 + 16 + h * 8);
    #pragma unroll
    for (int ct = 0; ct < 16; ++ct) {
      const _Float16* wrow = Wvh + (size_t)(ct * 16 + l16) * C_;
      AFrag af;
      af.h[0] = *(const v8h*)(wrow + c0 + h * 8);
      af.h[1] = *(const v8h*)(wrow + c0 + 16 + h * 8);
      acc[ct] = __builtin_amdgcn_wmma_f32_16x16x32_f16(
          false, af.v, false, bf.v, (short)0, acc[ct], false, false);
    }
  }

  // D tile: row M = co (h*8+v within ct-tile), col = n (l16) -> coalesced
  #pragma unroll
  for (int ct = 0; ct < 16; ++ct)
    #pragma unroll
    for (int v = 0; v < 8; ++v)
      Vc[((size_t)b * C_ + ct * 16 + h * 8 + v) * N_ + nt * 16 + l16] =
          (_Float16)acc[ct][v];
}

// ---------------------------------------------------------------------------
// Kernel 3: fused flash attention. One wave owns 16 query rows.
//   S = Qt[16,256] @ Kt^T (chunks of 32 keys), online softmax,
//   O += P(f16, via LDS transpose) @ Vc^T, epilogue O/l -> y [B,N,C] f32.
// ---------------------------------------------------------------------------
__global__ __launch_bounds__(256, 1) void attention_kernel(
    const _Float16* __restrict__ Qt,
    const _Float16* __restrict__ Kt,
    const _Float16* __restrict__ Vc,
    float* __restrict__ y) {
  // per-wave P transpose tile: 16 rows x 32 cols, row stride 40 halfs (80B,
  // keeps b128 alignment and spreads banks)
  __shared__ _Float16 plds[8][16 * 40];

  int lane = threadIdx.x & 31;
  int wv   = threadIdx.x >> 5;
  int l16  = lane & 15;
  int h    = lane >> 4;

  int w  = blockIdx.x * 8 + wv;   // 0..1023
  int b  = w >> 7;
  int qt = w & 127;
  int n0 = qt * 16;

  // Resident Q fragments (A layout), 8 x v16h
  AFrag qf[8];
  const _Float16* qrow = Qt + ((size_t)b * N_ + n0 + l16) * C_;
  #pragma unroll
  for (int kc = 0; kc < 8; ++kc) {
    qf[kc].h[0] = *(const v8h*)(qrow + kc * 32 + h * 8);
    qf[kc].h[1] = *(const v8h*)(qrow + kc * 32 + 16 + h * 8);
  }

  v8f o[16];
  #pragma unroll
  for (int t = 0; t < 16; ++t) o[t] = vzero8();
  float mrow[8], lrow[8];
  #pragma unroll
  for (int v = 0; v < 8; ++v) { mrow[v] = -3.0e38f; lrow[v] = 0.0f; }

  const _Float16* kbase = Kt + (size_t)b * N_ * C_;
  const _Float16* vbase = Vc + (size_t)b * (size_t)C_ * N_;
  _Float16* myl = plds[wv];

  #pragma unroll 1
  for (int m0 = 0; m0 < N_; m0 += 32) {
    if (m0 + 32 < N_)  // warm next K chunk (global_prefetch_b8)
      __builtin_prefetch(kbase + (size_t)(m0 + 32 + l16) * C_, 0, 0);

    // ---- S = Q @ K^T for 32 keys (two 16x16 tiles) ----
    v8f s0 = vzero8(), s1 = vzero8();
    #pragma unroll
    for (int kc = 0; kc < 8; ++kc) {
      int c0 = kc * 32;
      const _Float16* krow0 = kbase + (size_t)(m0 + l16) * C_ + c0 + h * 8;
      AFrag b0;
      b0.h[0] = *(const v8h*)(krow0);
      b0.h[1] = *(const v8h*)(krow0 + 16);
      s0 = __builtin_amdgcn_wmma_f32_16x16x32_f16(
          false, qf[kc].v, false, b0.v, (short)0, s0, false, false);
      const _Float16* krow1 = krow0 + (size_t)16 * C_;
      AFrag b1;
      b1.h[0] = *(const v8h*)(krow1);
      b1.h[1] = *(const v8h*)(krow1 + 16);
      s1 = __builtin_amdgcn_wmma_f32_16x16x32_f16(
          false, qf[kc].v, false, b1.v, (short)0, s1, false, false);
    }

    // ---- online softmax (rows live per (h, v); reduce across 16 lanes) ----
    float alpha[8];
    #pragma unroll
    for (int v = 0; v < 8; ++v) {
      float cm = fmaxf(s0[v], s1[v]);
      #pragma unroll
      for (int off = 1; off < 16; off <<= 1)
        cm = fmaxf(cm, __shfl_xor(cm, off, 16));
      float mnew = fmaxf(mrow[v], cm);
      alpha[v] = __expf(mrow[v] - mnew);
      mrow[v] = mnew;
      float p0 = __expf(s0[v] - mnew);
      float p1 = __expf(s1[v] - mnew);
      s0[v] = p0; s1[v] = p1;
      float rs = p0 + p1;
      #pragma unroll
      for (int off = 1; off < 16; off <<= 1)
        rs += __shfl_xor(rs, off, 16);
      lrow[v] = lrow[v] * alpha[v] + rs;
    }
    #pragma unroll
    for (int ct = 0; ct < 16; ++ct)
      #pragma unroll
      for (int v = 0; v < 8; ++v) o[ct][v] *= alpha[v];

    // ---- P: D-layout -> A-layout through LDS ----
    #pragma unroll
    for (int v = 0; v < 8; ++v) {
      myl[(h * 8 + v) * 40 + l16]      = (_Float16)s0[v];
      myl[(h * 8 + v) * 40 + 16 + l16] = (_Float16)s1[v];
    }
    asm volatile("s_wait_dscnt 0" ::: "memory");  // cross-lane RAW in LDS
    AFrag pf;
    pf.h[0] = *(const v8h*)(myl + l16 * 40 + h * 8);
    pf.h[1] = *(const v8h*)(myl + l16 * 40 + 16 + h * 8);

    // ---- O += P @ V^T over all 256 output channels ----
    #pragma unroll
    for (int ct = 0; ct < 16; ++ct) {
      const _Float16* vrow = vbase + (size_t)(ct * 16 + l16) * N_ + m0 + h * 8;
      AFrag vf;
      vf.h[0] = *(const v8h*)(vrow);
      vf.h[1] = *(const v8h*)(vrow + 16);
      o[ct] = __builtin_amdgcn_wmma_f32_16x16x32_f16(
          false, pf.v, false, vf.v, (short)0, o[ct], false, false);
    }
  }

  // ---- epilogue: divide by row sums, store y [B, N, C] f32 ----
  float linv[8];
  #pragma unroll
  for (int v = 0; v < 8; ++v) linv[v] = 1.0f / lrow[v];
  float* yrow = y + ((size_t)b * N_ + n0) * C_;
  #pragma unroll
  for (int ct = 0; ct < 16; ++ct)
    #pragma unroll
    for (int v = 0; v < 8; ++v)
      yrow[(size_t)(h * 8 + v) * C_ + ct * 16 + l16] = o[ct][v] * linv[v];
}

// ---------------------------------------------------------------------------
extern "C" void kernel_launch(void* const* d_in, const int* in_sizes, int n_in,
                              void* d_out, int out_size, void* d_ws, size_t ws_size,
                              hipStream_t stream) {
  (void)in_sizes; (void)n_in; (void)out_size; (void)ws_size;
  const float* x  = (const float*)d_in[0];
  const float* Wq = (const float*)d_in[1];
  const float* Wk = (const float*)d_in[2];
  const float* Wv = (const float*)d_in[3];
  float* y = (float*)d_out;

  char* ws = (char*)d_ws;
  _Float16* xT  = (_Float16*)(ws);                        // 8 MB
  _Float16* Qt  = (_Float16*)(ws + (size_t)(8u  << 20));  // 8 MB
  _Float16* Kt  = (_Float16*)(ws + (size_t)(16u << 20));  // 8 MB
  _Float16* Vc  = (_Float16*)(ws + (size_t)(24u << 20));  // 8 MB  ([C, N])
  _Float16* Wqh = (_Float16*)(ws + (size_t)(32u << 20));
  _Float16* Wkh = (_Float16*)(ws + (size_t)(32u << 20) + (128u << 10));
  _Float16* Wvh = (_Float16*)(ws + (size_t)(32u << 20) + (256u << 10));

  convert_w_kernel<<<(C_ * C_) / 256, 256, 0, stream>>>(Wq, Wk, Wv, Wqh, Wkh, Wvh);

  dim3 tgrid(N_ / 32, C_ / 32, B_);
  transpose_x_kernel<<<tgrid, dim3(32, 8), 0, stream>>>(x, xT);

  proj_qk_kernel<<<256, 256, 0, stream>>>(xT, Wqh, Wkh, Qt, Kt);
  proj_v_kernel<<<128, 256, 0, stream>>>(xT, Wvh, Vc);
  attention_kernel<<<128, 256, 0, stream>>>(Qt, Kt, Vc, y);
}
